// MultiHeadSelfAttentionBlock_67997922230530
// MI455X (gfx1250) — compile-verified, matrix operand order checked by reading, not run on hardware
//
#include <hip/hip_runtime.h>
#include <hip/hip_bf16.h>

typedef __bf16 bf16;
typedef __attribute__((ext_vector_type(8)))  bf16  v8bf;
typedef __attribute__((ext_vector_type(16))) bf16  v16bf;
typedef __attribute__((ext_vector_type(8)))  float v8f;

#define E_DIM 768
#define QKV_DIM 2304
#define SEQ 1024
#define BATCH 8
#define NHEAD 12
#define HDIM 64

// ---------------------------------------------------------------------------
// Fragment loader for V_WMMA_F32_16X16X32_BF16.
// A (16x32, MxK) per-lane layout (ISA 7.12.2): lanes 0-15 hold row M=lane with
// K=0..7 in VGPR0-3 and K=16..23 in VGPR4-7; lanes 16-31 hold row M=lane-16
// with K=8..15 / K=24..31. B (32x16, KxN) mirrors this when the B matrix is
// stored transposed [N,K] row-major. So for K-contiguous storage each lane
// does two 16-byte loads.
// ---------------------------------------------------------------------------
__device__ __forceinline__ v16bf load_frag(const bf16* p, int ld) {
  const int lane = threadIdx.x & 31;
  const bf16* q = p + (size_t)(lane & 15) * ld + ((lane >> 4) << 3);
  union { v16bf v; v8bf h[2]; } u;
  u.h[0] = *(const v8bf*)q;
  u.h[1] = *(const v8bf*)(q + 16);
  return u.v;
}

__device__ __forceinline__ v8f wmma_bf16(v16bf a, v16bf b, v8f c) {
  // (neg_a, A, neg_b, B, c_mod, C, reuse_a, reuse_b)
  return __builtin_amdgcn_wmma_f32_16x16x32_bf16(false, a, false, b, (short)0, c,
                                                 false, false);
}

// ---------------------------------------------------------------------------
// LayerNorm: one wave per row of 768, fused cast to bf16.
// ---------------------------------------------------------------------------
__global__ void ln_kernel(const float* __restrict__ x, const float* __restrict__ w,
                          const float* __restrict__ b, bf16* __restrict__ out) {
  const int row  = (blockIdx.x * blockDim.x + threadIdx.x) >> 5;
  const int lane = threadIdx.x & 31;
  const float* xr = x + (size_t)row * E_DIM;
  float s = 0.f, s2 = 0.f;
  for (int i = lane; i < E_DIM; i += 32) { float v = xr[i]; s += v; s2 += v * v; }
  for (int msk = 16; msk; msk >>= 1) { s += __shfl_xor(s, msk); s2 += __shfl_xor(s2, msk); }
  const float mu = s * (1.0f / E_DIM);
  const float var = s2 * (1.0f / E_DIM) - mu * mu;
  const float rs = rsqrtf(var + 1e-5f);
  bf16* o = out + (size_t)row * E_DIM;
  for (int i = lane; i < E_DIM; i += 32)
    o[i] = (bf16)((xr[i] - mu) * rs * w[i] + b[i]);
}

__global__ void cvt_kernel(const float* __restrict__ in, bf16* __restrict__ out, int n) {
  int i = blockIdx.x * blockDim.x + threadIdx.x;
  if (i < n) out[i] = (bf16)in[i];
}

// ---------------------------------------------------------------------------
// QKV GEMM: C[8192,2304] = xn[8192,768] * Wqkv[2304,768]^T + bias.
// One wave -> 64x64 tile (4x4 WMMA tiles): 16 WMMAs per 8 fragment loads per
// k-step. ~190 live VGPRs -> __launch_bounds__(256,1) lifts the default
// 128-VGPR occupancy cap so the accumulators stay in registers (no spills).
// Q written *1/8 into [B,H,S,D], K into [B,H,S,D], V transposed into
// [B,H,D,S] (so P*V is K-contiguous).
// ---------------------------------------------------------------------------
__global__ void __launch_bounds__(256, 1)
gemm_qkv_kernel(const bf16* __restrict__ A, const bf16* __restrict__ W,
                const float* __restrict__ bias,
                bf16* __restrict__ Qo, bf16* __restrict__ Ko,
                bf16* __restrict__ Vo) {
  const int lane = threadIdx.x & 31;
  const int wid = blockIdx.x * (blockDim.x >> 5) + (threadIdx.x >> 5);
  const int mt = wid / (QKV_DIM / 64);
  const int nt = wid % (QKV_DIM / 64);
  const int m0 = mt * 64, n0 = nt * 64;
  const bf16* Ap = A + (size_t)m0 * E_DIM;
  const bf16* Wp = W + (size_t)n0 * E_DIM;

  v8f acc[4][4] = {};
  for (int k = 0; k < E_DIM; k += 32) {
    __builtin_prefetch(Ap + k + 256, 0, 1);
    v16bf a[4];
#pragma unroll
    for (int t = 0; t < 4; ++t)
      a[t] = load_frag(Ap + (size_t)(16 * t) * E_DIM + k, E_DIM);
#pragma unroll
    for (int j = 0; j < 4; ++j) {
      v16bf bj = load_frag(Wp + (size_t)(16 * j) * E_DIM + k, E_DIM);
#pragma unroll
      for (int t = 0; t < 4; ++t)
        acc[t][j] = wmma_bf16(a[t], bj, acc[t][j]);
    }
  }

  // C layout: VGPR i -> row M = i + 8*(lane>=16); column N = lane & 15.
  const int col = lane & 15;
  const int rbase = (lane >> 4) << 3;
#pragma unroll
  for (int t = 0; t < 4; ++t) {
#pragma unroll
    for (int j = 0; j < 4; ++j) {
      const int nn = n0 + j * 16 + col;
      const float bv = bias[nn];
      const int sec = nn / E_DIM;          // 0=Q 1=K 2=V
      const int e = nn % E_DIM;
      const int h = e >> 6, d = e & 63;
#pragma unroll
      for (int i = 0; i < 8; ++i) {
        const int mm = m0 + t * 16 + rbase + i;
        const int bb = mm >> 10, ss = mm & (SEQ - 1);
        const float v = acc[t][j][i] + bv;
        const size_t bh = (size_t)bb * NHEAD + h;
        if (sec == 0)      Qo[(bh * SEQ + ss) * HDIM + d] = (bf16)(v * 0.125f);
        else if (sec == 1) Ko[(bh * SEQ + ss) * HDIM + d] = (bf16)v;
        else               Vo[(bh * HDIM + d) * SEQ + ss] = (bf16)v;
      }
    }
  }
}

// ---------------------------------------------------------------------------
// Flash-style attention. One wave per 16-query tile of one (b,h).
// Streams 32 keys per iteration: 4 WMMAs scores, online softmax, P staged
// through LDS (C-layout -> A-fragment relayout), 4 WMMAs ctx. K/V per head is
// 256KB -> L2-resident across the 64 query tiles that share it.
// ---------------------------------------------------------------------------
__global__ void __launch_bounds__(128, 1)
attn_kernel(const bf16* __restrict__ Q, const bf16* __restrict__ K,
            const bf16* __restrict__ Vt, bf16* __restrict__ ctx) {
  const int lane = threadIdx.x & 31;
  const int wv = threadIdx.x >> 5;                 // 0..3
  const int tile = blockIdx.x * 4 + wv;            // 0..6143
  const int bh = tile >> 6;                        // (b,h) index
  const int qt = tile & 63;                        // query tile in sequence
  const int b = bh / NHEAD, h = bh % NHEAD;

  __shared__ bf16 pbuf[4][16][32];
  bf16 (*P)[32] = pbuf[wv];

  const bf16* Qp = Q + ((size_t)bh * SEQ + qt * 16) * HDIM;
  const bf16* Kp = K + (size_t)bh * SEQ * HDIM;
  const bf16* Vp = Vt + (size_t)bh * HDIM * SEQ;

  const v16bf qa0 = load_frag(Qp, HDIM);           // d = 0..31  (pre-scaled)
  const v16bf qa1 = load_frag(Qp + 32, HDIM);      // d = 32..63

  v8f acc[4] = {};                                 // ctx, 4 d-tiles of 16
  float m[8], l[8];
#pragma unroll
  for (int i = 0; i < 8; ++i) { m[i] = -3.0e38f; l[i] = 0.f; }

  const int col = lane & 15;
  const int rbase = (lane >> 4) << 3;

  for (int j = 0; j < SEQ; j += 32) {
    v8f s0 = {}, s1 = {};
    s0 = wmma_bf16(qa0, load_frag(Kp + (size_t)j * HDIM, HDIM), s0);
    s0 = wmma_bf16(qa1, load_frag(Kp + (size_t)j * HDIM + 32, HDIM), s0);
    s1 = wmma_bf16(qa0, load_frag(Kp + (size_t)(j + 16) * HDIM, HDIM), s1);
    s1 = wmma_bf16(qa1, load_frag(Kp + (size_t)(j + 16) * HDIM + 32, HDIM), s1);

    float corr[8];
#pragma unroll
    for (int i = 0; i < 8; ++i) {
      // row i+rbase lives entirely in this 16-lane half -> shfl_xor <= 8 stays inside
      float smax = fmaxf(s0[i], s1[i]);
      smax = fmaxf(smax, __shfl_xor(smax, 1));
      smax = fmaxf(smax, __shfl_xor(smax, 2));
      smax = fmaxf(smax, __shfl_xor(smax, 4));
      smax = fmaxf(smax, __shfl_xor(smax, 8));
      const float mnew = fmaxf(m[i], smax);
      const float c = __expf(m[i] - mnew);
      const float p0 = __expf(s0[i] - mnew);
      const float p1 = __expf(s1[i] - mnew);
      float rsum = p0 + p1;
      rsum += __shfl_xor(rsum, 1);
      rsum += __shfl_xor(rsum, 2);
      rsum += __shfl_xor(rsum, 4);
      rsum += __shfl_xor(rsum, 8);
      l[i] = l[i] * c + rsum;
      m[i] = mnew;
      corr[i] = c;
      P[i + rbase][col]      = (bf16)p0;
      P[i + rbase][col + 16] = (bf16)p1;
    }
#pragma unroll
    for (int n = 0; n < 4; ++n)
#pragma unroll
      for (int i = 0; i < 8; ++i) acc[n][i] *= corr[i];

    // wave-local LDS relayout: wait for the ds stores from all 32 lanes
    asm volatile("s_wait_dscnt 0x0" ::: "memory");
    const v16bf pf = load_frag(&P[0][0], 32);
#pragma unroll
    for (int n = 0; n < 4; ++n)
      acc[n] = wmma_bf16(pf, load_frag(Vp + (size_t)(n * 16) * SEQ + j, SEQ), acc[n]);
  }

  // normalize and scatter to ctx[B,S,E] bf16 for the out-projection GEMM
#pragma unroll
  for (int n = 0; n < 4; ++n)
#pragma unroll
    for (int i = 0; i < 8; ++i) {
      const int srow = qt * 16 + rbase + i;
      const int cg = h * HDIM + n * 16 + col;
      ctx[((size_t)b * SEQ + srow) * E_DIM + cg] = (bf16)(acc[n][i] / l[i]);
    }
}

// ---------------------------------------------------------------------------
// Output projection: out[8192,768] = ctx[8192,768] * Wout[768,768]^T + bias (f32)
// One wave -> 64x64 tile, same structure as the QKV GEMM.
// ---------------------------------------------------------------------------
__global__ void __launch_bounds__(256, 1)
gemm_out_kernel(const bf16* __restrict__ A, const bf16* __restrict__ W,
                const float* __restrict__ bias, float* __restrict__ out) {
  const int lane = threadIdx.x & 31;
  const int wid = blockIdx.x * (blockDim.x >> 5) + (threadIdx.x >> 5);
  const int mt = wid / (E_DIM / 64);
  const int nt = wid % (E_DIM / 64);
  const int m0 = mt * 64, n0 = nt * 64;
  const bf16* Ap = A + (size_t)m0 * E_DIM;
  const bf16* Wp = W + (size_t)n0 * E_DIM;

  v8f acc[4][4] = {};
  for (int k = 0; k < E_DIM; k += 32) {
    __builtin_prefetch(Ap + k + 256, 0, 1);
    v16bf a[4];
#pragma unroll
    for (int t = 0; t < 4; ++t)
      a[t] = load_frag(Ap + (size_t)(16 * t) * E_DIM + k, E_DIM);
#pragma unroll
    for (int j = 0; j < 4; ++j) {
      v16bf bj = load_frag(Wp + (size_t)(16 * j) * E_DIM + k, E_DIM);
#pragma unroll
      for (int t = 0; t < 4; ++t)
        acc[t][j] = wmma_bf16(a[t], bj, acc[t][j]);
    }
  }

  const int col = lane & 15;
  const int rbase = (lane >> 4) << 3;
#pragma unroll
  for (int t = 0; t < 4; ++t)
#pragma unroll
    for (int j = 0; j < 4; ++j) {
      const int nn = n0 + j * 16 + col;
      const float bv = bias[nn];
#pragma unroll
      for (int i = 0; i < 8; ++i) {
        const int mm = m0 + t * 16 + rbase + i;
        out[(size_t)mm * E_DIM + nn] = acc[t][j][i] + bv;
      }
    }
}

// ---------------------------------------------------------------------------
extern "C" void kernel_launch(void* const* d_in, const int* in_sizes, int n_in,
                              void* d_out, int out_size, void* d_ws, size_t ws_size,
                              hipStream_t stream) {
  (void)in_sizes; (void)n_in; (void)out_size; (void)ws_size;
  const float* x      = (const float*)d_in[0];
  const float* ln_w   = (const float*)d_in[1];
  const float* ln_b   = (const float*)d_in[2];
  const float* wqkv_f = (const float*)d_in[3];
  const float* bqkv   = (const float*)d_in[4];
  const float* wout_f = (const float*)d_in[5];
  const float* bout   = (const float*)d_in[6];
  float* out = (float*)d_out;

  const size_t rows = (size_t)BATCH * SEQ;           // 8192
  const size_t qkvElems = (size_t)BATCH * NHEAD * SEQ * HDIM;  // 6,291,456

  bf16* xn   = (bf16*)d_ws;                          // [8192,768]
  bf16* wqkv = xn + rows * E_DIM;                    // [2304,768]
  bf16* wout = wqkv + (size_t)QKV_DIM * E_DIM;       // [768,768]
  bf16* Qb   = wout + (size_t)E_DIM * E_DIM;         // [B,H,S,D]
  bf16* Kb   = Qb + qkvElems;                        // [B,H,S,D]
  bf16* Vb   = Kb + qkvElems;                        // [B,H,D,S] (transposed)
  bf16* ctxb = Vb + qkvElems;                        // [8192,768]

  // 1) LayerNorm + cast (8 rows per 256-thread block)
  ln_kernel<<<(int)(rows / 8), 256, 0, stream>>>(x, ln_w, ln_b, xn);

  // 2) weight casts
  cvt_kernel<<<(QKV_DIM * E_DIM + 255) / 256, 256, 0, stream>>>(wqkv_f, wqkv, QKV_DIM * E_DIM);
  cvt_kernel<<<(E_DIM * E_DIM + 255) / 256, 256, 0, stream>>>(wout_f, wout, E_DIM * E_DIM);

  // 3) QKV projection: (8192/64) * (2304/64) = 4608 waves, 8 waves/block
  gemm_qkv_kernel<<<4608 / 8, 256, 0, stream>>>(xn, wqkv, bqkv, Qb, Kb, Vb);

  // 4) attention: B*H*(S/16) = 6144 tiles, 4 waves/block
  attn_kernel<<<6144 / 4, 128, 0, stream>>>(Qb, Kb, Vb, ctxb);

  // 5) output projection: (8192/64) * (768/64) = 1536 waves, 8 waves/block
  gemm_out_kernel<<<1536 / 8, 256, 0, stream>>>(ctxb, wout, bout, out);
}
